// RGCNSampling_1520418423099
// MI455X (gfx1250) — matrix operand order repaired and checked by reading, not compiled, not run on hardware
//
#include <hip/hip_runtime.h>
#include <hip/hip_bf16.h>

#define N_NODES 100000
#define N_REL   8
#define N_EDGES 400000
#define DIM     128
#define N_LAYERS 3
#define ND      (N_NODES * DIM)
#define M_TILES (N_NODES / 16)     // 6250
#define GEMM_WAVES_PER_BLOCK 10    // 320 threads; 625 * 10 = 6250 exactly

typedef __attribute__((ext_vector_type(16))) __bf16 v16bf;
typedef __attribute__((ext_vector_type(8)))  __bf16 v8bf;
typedef __attribute__((ext_vector_type(8)))  float  v8f;

// ---------- helpers ----------
__device__ __forceinline__ unsigned short f2bf(float f) {
    unsigned b = __float_as_uint(f);
    unsigned r = b + 0x7FFFu + ((b >> 16) & 1u);   // round-to-nearest-even
    return (unsigned short)(r >> 16);
}
__device__ __forceinline__ unsigned f2ord(float f) {   // order-preserving f32->u32
    unsigned b = __float_as_uint(f);
    return (b & 0x80000000u) ? ~b : (b | 0x80000000u);
}
__device__ __forceinline__ float ord2f(unsigned u) {
    unsigned b = (u & 0x80000000u) ? (u ^ 0x80000000u) : ~u;
    return __uint_as_float(b);
}

// ---------- small prep kernels ----------
__global__ void k_wconv(const float* __restrict__ w, unsigned short* __restrict__ wb, int n) {
    int t = blockIdx.x * blockDim.x + threadIdx.x;
    if (t < n) wb[t] = f2bf(w[t]);
}
__global__ void k_xconv(const float* __restrict__ x, unsigned short* __restrict__ xb) {
    int t = blockIdx.x * blockDim.x + threadIdx.x;
    if (t < ND) xb[t] = f2bf(x[t]);
}
// vl[k] = sum_j Wsrc[j,k]*al[j] ; vr[k] = sum_j Wdst[j,k]*ar[j]   (all layers*rels)
__global__ void k_vlvr(const float* __restrict__ Ws, const float* __restrict__ Wd,
                       const float* __restrict__ al, const float* __restrict__ ar,
                       float* __restrict__ vl, float* __restrict__ vr) {
    int lr = blockIdx.x;            // 0..23
    int k  = threadIdx.x;           // 0..127
    const float* ws = Ws + lr * DIM * DIM;
    const float* wd = Wd + lr * DIM * DIM;
    const float* a0 = al + lr * DIM;
    const float* a1 = ar + lr * DIM;
    float sl = 0.f, sr = 0.f;
    for (int j = 0; j < DIM; ++j) {
        sl += ws[j * DIM + k] * a0[j];
        sr += wd[j * DIM + k] * a1[j];
    }
    vl[lr * DIM + k] = sl;
    vr[lr * DIM + k] = sr;
}
__global__ void k_bias_init(float* __restrict__ out, const float* __restrict__ bias) {
    int t = blockIdx.x * blockDim.x + threadIdx.x;
    if (t < ND) out[t] = bias[t & (DIM - 1)];
}
__global__ void k_relu(float* __restrict__ x) {
    int t = blockIdx.x * blockDim.x + threadIdx.x;
    if (t < ND) x[t] = fmaxf(x[t], 0.f);
}
__global__ void k_reset(unsigned* __restrict__ emax_u, float* __restrict__ denom) {
    int t = blockIdx.x * blockDim.x + threadIdx.x;
    if (t < N_NODES) { emax_u[t] = f2ord(-__builtin_inff()); denom[t] = 0.f; }
}

// ---------- WMMA GEMM: zs[100000,128] = x_bf16 @ W_bf16^T ----------
// B (128x128 bf16 weight tile, 32KB) is staged in LDS ONCE per workgroup,
// pre-permuted into the per-lane WMMA fragment layout: fragment f=(nt,kc,lane)
// occupies 16 contiguous bf16 (32B), consecutive lanes at consecutive 32B ->
// conflict-free ds_load_b128 reads. One wave computes a full 16(M)x128(N)
// row-block: A fragments loaded once, 32 WMMAs per wave. Grid divides exactly
// (625 blocks x 10 waves = 6250 = M_TILES) so EXEC is all-ones everywhere and
// the barrier is uniform.
__global__ __launch_bounds__(32 * GEMM_WAVES_PER_BLOCK)
void k_gemm_zs(const unsigned short* __restrict__ xb,
               const unsigned short* __restrict__ wb,
               float* __restrict__ zs) {
    __shared__ unsigned short sB[8 * 4 * 32 * 16];   // 16384 elems = 32KB

    const int lane = threadIdx.x & 31;
    const int row  = lane & 15;
    const int hi   = lane >> 4;

    // ---- stage B into LDS in fragment-major layout ----
    // fragment f: nt=f>>7, kc=(f>>5)&3, frag_lane=f&31
    // global source: row (nt*16 + (frag_lane&15)) of W, cols kc*32 + (frag_lane>>4)*16 .. +15
    for (int f = threadIdx.x; f < 1024; f += 32 * GEMM_WAVES_PER_BLOCK) {
        const int fnt = f >> 7, fkc = (f >> 5) & 3, fl = f & 31;
        const size_t goff = (size_t)(fnt * 16 + (fl & 15)) * DIM + fkc * 32 + (fl >> 4) * 16;
        *(v16bf*)(sB + f * 16) = *(const v16bf*)(wb + goff);
    }
    __syncthreads();

    const int mtile = blockIdx.x * GEMM_WAVES_PER_BLOCK + (threadIdx.x >> 5); // 0..6249
    const int mbase = mtile * 16;
    const unsigned short* arow = xb + (size_t)(mbase + row) * DIM;

    // A 16x32 fragments: lanes 0-15 hold K {0..15}, lanes 16-31 hold K {16..31}
    v16bf a[4];
#pragma unroll
    for (int kc = 0; kc < 4; ++kc) {
        a[kc] = *(const v16bf*)(arow + kc * 32 + hi * 16);
    }

#pragma unroll
    for (int nt = 0; nt < 8; ++nt) {
        v8f c = {};
#pragma unroll
        for (int kc = 0; kc < 4; ++kc) {
            v16bf b = *(const v16bf*)(sB + ((nt * 4 + kc) * 32 + lane) * 16);
            c = __builtin_amdgcn_wmma_f32_16x16x32_bf16(false, a[kc], false, b,
                                                        (short)0, c, false, false);
        }
        // D 16x16 f32: VGPR v -> M = v + 8*hi ; N = lane&15
#pragma unroll
        for (int v = 0; v < 8; ++v)
            zs[(size_t)(mbase + v + hi * 8) * DIM + nt * 16 + row] = c[v];
    }
}

// ---------- per-node attention scalars: el = x.vl, er = x.vr ----------
__global__ __launch_bounds__(256) void k_node_attn(const float* __restrict__ x,
                                                   const float* __restrict__ vl,
                                                   const float* __restrict__ vr,
                                                   float* __restrict__ el,
                                                   float* __restrict__ er) {
    int wid  = (blockIdx.x * blockDim.x + threadIdx.x) >> 5;
    int lane = threadIdx.x & 31;
    if (wid >= N_NODES) return;
    const float* xr = x + (size_t)wid * DIM;
    float pl = 0.f, pr = 0.f;
#pragma unroll
    for (int j = 0; j < 4; ++j) {
        float xv = xr[lane + 32 * j];
        pl += xv * vl[lane + 32 * j];
        pr += xv * vr[lane + 32 * j];
    }
#pragma unroll
    for (int off = 16; off; off >>= 1) {
        pl += __shfl_xor(pl, off);
        pr += __shfl_xor(pr, off);
    }
    if (lane == 0) { el[wid] = pl; er[wid] = pr; }
}

// ---------- edge pass A: e = lrelu(el[src]+er[dst]); segment max ----------
__global__ void k_edge_a(const int* __restrict__ src, const int* __restrict__ dst,
                         const float* __restrict__ el, const float* __restrict__ er,
                         float* __restrict__ eval, unsigned* __restrict__ emax_u) {
    int i = blockIdx.x * blockDim.x + threadIdx.x;
    if (i >= N_EDGES) return;
    float e = el[src[i]] + er[dst[i]];
    e = e > 0.f ? e : 0.2f * e;
    eval[i] = e;
    atomicMax(&emax_u[dst[i]], f2ord(e));
}

// ---------- edge pass B: w = exp(e - max); segment sum ----------
__global__ void k_edge_b(const int* __restrict__ dst, float* __restrict__ eval,
                         const unsigned* __restrict__ emax_u, float* __restrict__ denom) {
    int i = blockIdx.x * blockDim.x + threadIdx.x;
    if (i >= N_EDGES) return;
    int d = dst[i];
    float w = __expf(eval[i] - ord2f(emax_u[d]));
    eval[i] = w;
    atomicAdd(&denom[d], w);
}

// ---------- edge pass C: out[dst] += zs[src] * alpha (one wave per edge) ----------
__global__ __launch_bounds__(256) void k_edge_c(const int* __restrict__ src,
                                                const int* __restrict__ dst,
                                                const float* __restrict__ eval,
                                                const float* __restrict__ denom,
                                                const float* __restrict__ zs,
                                                float* __restrict__ out) {
    int wid  = (blockIdx.x * blockDim.x + threadIdx.x) >> 5;
    int lane = threadIdx.x & 31;
    if (wid >= N_EDGES) return;
    int s = src[wid], d = dst[wid];
    float alpha = eval[wid] / denom[d];
    const float4 z = *(const float4*)(zs + (size_t)s * DIM + lane * 4);
    float* o = out + (size_t)d * DIM + lane * 4;
    atomicAdd(o + 0, z.x * alpha);
    atomicAdd(o + 1, z.y * alpha);
    atomicAdd(o + 2, z.z * alpha);
    atomicAdd(o + 3, z.w * alpha);
}

// ---------- launch ----------
extern "C" void kernel_launch(void* const* d_in, const int* in_sizes, int n_in,
                              void* d_out, int out_size, void* d_ws, size_t ws_size,
                              hipStream_t stream) {
    const float* h    = (const float*)d_in[0];
    const int*   esrc = (const int*)d_in[1];
    const int*   edst = (const int*)d_in[2];
    const float* fcs  = (const float*)d_in[3];
    const float* fcd  = (const float*)d_in[4];
    const float* al   = (const float*)d_in[5];
    const float* ar   = (const float*)d_in[6];
    const float* hb   = (const float*)d_in[7];
    float* out_final  = (float*)d_out;

    // workspace partition (256B aligned)
    char* p = (char*)d_ws;
    auto take = [&](size_t bytes) {
        char* r = p;
        p += (bytes + 255) & ~(size_t)255;
        return r;
    };
    float*          x1     = (float*)take(sizeof(float) * ND);
    float*          x2     = (float*)take(sizeof(float) * ND);
    float*          zs     = (float*)take(sizeof(float) * ND);
    unsigned short* xb     = (unsigned short*)take(sizeof(unsigned short) * ND);
    unsigned short* wb     = (unsigned short*)take(sizeof(unsigned short) * N_LAYERS * N_REL * DIM * DIM);
    float*          vl     = (float*)take(sizeof(float) * N_LAYERS * N_REL * DIM);
    float*          vr     = (float*)take(sizeof(float) * N_LAYERS * N_REL * DIM);
    float*          el     = (float*)take(sizeof(float) * N_NODES);
    float*          er     = (float*)take(sizeof(float) * N_NODES);
    unsigned*       emax_u = (unsigned*)take(sizeof(unsigned) * N_NODES);
    float*          denom  = (float*)take(sizeof(float) * N_NODES);
    float*          eval   = (float*)take(sizeof(float) * N_EDGES);

    const int NW = N_LAYERS * N_REL * DIM * DIM;   // 393216
    k_wconv<<<(NW + 255) / 256, 256, 0, stream>>>(fcs, wb, NW);
    k_vlvr<<<N_LAYERS * N_REL, DIM, 0, stream>>>(fcs, fcd, al, ar, vl, vr);

    const float* xcur = h;
    float* layer_out[N_LAYERS] = { x1, x2, out_final };

    for (int l = 0; l < N_LAYERS; ++l) {
        float* xout = layer_out[l];
        k_xconv<<<ND / 256, 256, 0, stream>>>(xcur, xb);
        k_bias_init<<<ND / 256, 256, 0, stream>>>(xout, hb + l * DIM);
        for (int r = 0; r < N_REL; ++r) {
            const int lr = l * N_REL + r;
            const int* sr = esrc + (size_t)r * N_EDGES;
            const int* ds = edst + (size_t)r * N_EDGES;
            k_gemm_zs<<<M_TILES / GEMM_WAVES_PER_BLOCK, 32 * GEMM_WAVES_PER_BLOCK, 0, stream>>>(
                xb, wb + (size_t)lr * DIM * DIM, zs);
            k_node_attn<<<(N_NODES + 7) / 8, 256, 0, stream>>>(xcur, vl + lr * DIM, vr + lr * DIM, el, er);
            k_reset<<<(N_NODES + 255) / 256, 256, 0, stream>>>(emax_u, denom);
            k_edge_a<<<(N_EDGES + 255) / 256, 256, 0, stream>>>(sr, ds, el, er, eval, emax_u);
            k_edge_b<<<(N_EDGES + 255) / 256, 256, 0, stream>>>(ds, eval, emax_u, denom);
            k_edge_c<<<N_EDGES / 8, 256, 0, stream>>>(sr, ds, eval, denom, zs, xout);
        }
        if (l < N_LAYERS - 1) k_relu<<<ND / 256, 256, 0, stream>>>(xout);
        xcur = xout;
    }
}
